// Sequence_56968446214710
// MI455X (gfx1250) — matrix operand order repaired and checked by reading, not compiled, not run on hardware
//
#include <hip/hip_runtime.h>
#include <hip/hip_bf16.h>

// ---------------- problem constants (match reference) ----------------
constexpr int HID  = 512;     // hidden size
constexpr int BAT  = 64;      // batch
constexpr int NWG  = 32;      // workgroups: each owns 16 hidden units
constexpr int TPB  = 128;     // 4 waves of 32: wave w owns gate w (i/f/g/o)

typedef __bf16 bf16_t;
typedef __attribute__((ext_vector_type(8)))  __bf16 v8bf;
typedef __attribute__((ext_vector_type(16))) __bf16 v16bf;
typedef __attribute__((ext_vector_type(8)))  float  v8f;

// ---------------- workspace layout (bytes) ----------------
constexpr size_t OFF_BAR  = 0;                               // 2 uints (count, gen)
constexpr size_t OFF_PART = 256;                             // NWG*BAT floats (head partials)
constexpr size_t OFF_H1   = 16384;                           // bf16 [2][BAT*HID] ping-pong
constexpr size_t OFF_H2   = OFF_H1 + 2ull * BAT * HID * 2;   // bf16 [2][BAT*HID]
constexpr size_t OFF_W    = OFF_H2 + 2ull * BAT * HID * 2;   // bf16 [3][2048][512] (U1,W2,U2)
constexpr size_t WMAT     = 2048ull * 512;                   // elements per matrix

// ---------------- grid-wide barrier (deterministic spin) ----------------
__device__ __forceinline__ void grid_barrier(unsigned* cnt, unsigned* gen) {
    __syncthreads();
    if (threadIdx.x == 0) {
        __threadfence();
        unsigned g = __hip_atomic_load(gen, __ATOMIC_RELAXED, __HIP_MEMORY_SCOPE_AGENT);
        unsigned a = __hip_atomic_fetch_add(cnt, 1u, __ATOMIC_ACQ_REL, __HIP_MEMORY_SCOPE_AGENT);
        if (a == NWG - 1) {
            __hip_atomic_store(cnt, 0u, __ATOMIC_RELAXED, __HIP_MEMORY_SCOPE_AGENT);
            __hip_atomic_fetch_add(gen, 1u, __ATOMIC_RELEASE, __HIP_MEMORY_SCOPE_AGENT);
        } else {
            while (__hip_atomic_load(gen, __ATOMIC_ACQUIRE, __HIP_MEMORY_SCOPE_AGENT) == g) {
                __builtin_amdgcn_s_sleep(1);
            }
        }
    }
    __syncthreads();
}

// A-fragment (contiguous k order, matching B): lane l holds row M=l%16,
// K = koffA + {0..15}   (koffA = lane<16 ? 0 : 16)
__device__ __forceinline__ v16bf load_a(const bf16_t* __restrict__ base, int row, int k) {
    v8bf lo = *(const v8bf*)(base + row * HID + k);
    v8bf hi = *(const v8bf*)(base + row * HID + k + 8);
    return __builtin_shufflevector(lo, hi, 0,1,2,3,4,5,6,7,8,9,10,11,12,13,14,15);
}

// ---------------- prep: f32->bf16 weights, zero state ----------------
__global__ void prep_kernel(const float* __restrict__ U1, const float* __restrict__ W2,
                            const float* __restrict__ U2, unsigned char* __restrict__ ws) {
    size_t i = (size_t)blockIdx.x * blockDim.x + threadIdx.x;
    bf16_t* wbf = (bf16_t*)(ws + OFF_W);
    if (i < 3 * WMAT) {
        const float* src = (i < WMAT) ? U1 : (i < 2 * WMAT ? W2 : U2);
        wbf[i] = (bf16_t)src[i % WMAT];
    }
    bf16_t* h = (bf16_t*)(ws + OFF_H1);          // h1[2] + h2[2] contiguous
    if (i < 4ull * BAT * HID) h[i] = (bf16_t)0.f;
    if (i < (size_t)(NWG * BAT)) ((float*)(ws + OFF_PART))[i] = 0.f;
    if (i < 2) ((unsigned*)(ws + OFF_BAR))[i] = 0u;
}

// ---------------- persistent recurrence kernel ----------------
__global__ __launch_bounds__(TPB, 1) void lstm_kernel(
        const float* __restrict__ x,  const float* __restrict__ W1,
        const float* __restrict__ bW1, const float* __restrict__ bU1,
        const float* __restrict__ bW2, const float* __restrict__ bU2,
        const float* __restrict__ lw,  const float* __restrict__ lb,
        const int* __restrict__ fut,   float* __restrict__ out,
        unsigned char* __restrict__ ws, int SEQ) {

    // LDS: weight slices resident for the whole sequence + per-step staging
    __shared__ __align__(32) bf16_t Wlds[3][4][16][HID];   // 192 KB (U1,W2,U2 slices)
    __shared__ float gLDS[4][BAT][16];                     // 16 KB raw gates
    __shared__ float cst[2][BAT][16];                      // c1,c2 (f32 state)
    __shared__ float htile[BAT][16];                       // h2 tile for head
    __shared__ float xt[BAT];
    __shared__ float b1s[4][16], b2s[4][16], w1s[4][16], lws[16];

    const int tid  = threadIdx.x;
    const int lane = tid & 31;
    const int wv   = tid >> 5;          // gate index 0..3
    const int jb   = blockIdx.x * 16;   // this WG's hidden-unit slice

    // per-WG constants
    if (tid < 64) {
        int g = tid >> 4, r = tid & 15;
        int row = g * HID + jb + r;     // row of the [4H, *] matrices
        b1s[g][r] = bW1[row] + bU1[row];
        b2s[g][r] = bW2[row] + bU2[row];
        w1s[g][r] = W1[row];
    }
    if (tid < 16) lws[tid] = lw[jb + tid];
    for (int e = tid; e < 2 * BAT * 16; e += TPB) ((float*)cst)[e] = 0.f;

    // stage weight slices into LDS once via the CDNA5 async global->LDS path:
    // GLOBAL_LOAD_ASYNC_TO_LDS_B128 moves 16B/lane directly into LDS with no
    // VGPR round-trip; drained with s_wait_asynccnt before compute.
    {
        const bf16_t* wg = (const bf16_t*)(ws + OFF_W);
        constexpr int CH = HID / 8;                      // 16B chunks per row
        for (int idx = tid; idx < 3 * 4 * 16 * CH; idx += TPB) {
            int chunk = idx % CH;
            int row16 = idx / CH;                        // 0..191
            int r = row16 & 15, g = (row16 >> 4) & 3, mat = row16 >> 6;
            int grow = g * HID + jb + r;
            unsigned lds_off = (unsigned)(uintptr_t)&Wlds[mat][g][r][chunk * 8];
            unsigned long long ga =
                (unsigned long long)(uintptr_t)&wg[((size_t)mat * 2048 + grow) * HID + chunk * 8];
            asm volatile("global_load_async_to_lds_b128 %0, %1, off"
                         :: "v"(lds_off), "v"(ga) : "memory");
        }
        asm volatile("s_wait_asynccnt 0x0" ::: "memory");
    }
    __syncthreads();

    bf16_t*  h1b  = (bf16_t*)(ws + OFF_H1);
    bf16_t*  h2b  = (bf16_t*)(ws + OFF_H2);
    float*   part = (float*)(ws + OFF_PART);
    unsigned* bcnt = (unsigned*)(ws + OFF_BAR);
    unsigned* bgen = bcnt + 1;

    const float lbv = lb[0];
    const int F = fut[0];
    const int T = SEQ + F;
    const int mrow  = lane & 15;
    const int koffA = (lane < 16) ? 0 : 16;
    const int koffB = (lane < 16) ? 0 : 16;
    const int bofs  = (lane < 16) ? 0 : 8;
    const v8f vzero = {0.f, 0.f, 0.f, 0.f, 0.f, 0.f, 0.f, 0.f};

    for (int t = 0; t < T; ++t) {
        // ---- x_t + previous-step output writeback (fixed-order reduce) ----
        if (tid < BAT) {
            float red = lbv;
            #pragma unroll 4
            for (int w = 0; w < NWG; ++w) red += part[w * BAT + tid];
            xt[tid] = (t < SEQ) ? x[tid * SEQ + t] : red;
            if (blockIdx.x == 0 && t > 0) out[tid * T + (t - 1)] = red;
        }
        __syncthreads();

        // ================= cell 1: gates = x*W1^T + b1 + h1 @ U1^T =================
        const bf16_t* hp1 = h1b + ((t + 1) & 1) * (BAT * HID);
        bf16_t*       hn1 = h1b + (t & 1) * (BAT * HID);
        {
            v8f acc[4];
            #pragma unroll
            for (int mt = 0; mt < 4; ++mt) acc[mt] = vzero;
            for (int kk = 0; kk < HID; kk += 32) {
                v16bf bfrag = *(const v16bf*)&Wlds[0][wv][mrow][kk + koffB];
                #pragma unroll
                for (int mt = 0; mt < 4; ++mt) {
                    v16bf afrag = load_a(hp1, mt * 16 + mrow, kk + koffA);
                    acc[mt] = __builtin_amdgcn_wmma_f32_16x16x32_bf16(
                        false, afrag, false, bfrag, (short)0, acc[mt], false, false);
                }
            }
            const float bias = b1s[wv][mrow], w1v = w1s[wv][mrow];
            #pragma unroll
            for (int mt = 0; mt < 4; ++mt)
                #pragma unroll
                for (int r = 0; r < 8; ++r) {
                    int bb = mt * 16 + r + bofs;
                    gLDS[wv][bb][mrow] = acc[mt][r] + bias + xt[bb] * w1v;
                }
        }
        __syncthreads();
        for (int e = tid; e < BAT * 16; e += TPB) {      // un-squashed LSTM cell
            int bb = e >> 4, j = e & 15;
            float i_ = gLDS[0][bb][j], f_ = gLDS[1][bb][j];
            float g_ = gLDS[2][bb][j], o_ = gLDS[3][bb][j];
            float c = f_ * cst[0][bb][j] + i_ * g_;
            cst[0][bb][j] = c;
            hn1[bb * HID + jb + j] = (bf16_t)(o_ * tanhf(c));
        }
        grid_barrier(bcnt, bgen);                        // full h1_t visible

        // ============ cell 2: gates = h1_t @ W2^T + h2 @ U2^T + b2 ============
        const bf16_t* hp2 = h2b + ((t + 1) & 1) * (BAT * HID);
        bf16_t*       hn2 = h2b + (t & 1) * (BAT * HID);
        {
            v8f acc[4];
            #pragma unroll
            for (int mt = 0; mt < 4; ++mt) acc[mt] = vzero;
            for (int kk = 0; kk < HID; kk += 32) {
                v16bf bfrag = *(const v16bf*)&Wlds[1][wv][mrow][kk + koffB];   // W2
                #pragma unroll
                for (int mt = 0; mt < 4; ++mt) {
                    v16bf afrag = load_a(hn1, mt * 16 + mrow, kk + koffA);
                    acc[mt] = __builtin_amdgcn_wmma_f32_16x16x32_bf16(
                        false, afrag, false, bfrag, (short)0, acc[mt], false, false);
                }
            }
            for (int kk = 0; kk < HID; kk += 32) {
                v16bf bfrag = *(const v16bf*)&Wlds[2][wv][mrow][kk + koffB];   // U2
                #pragma unroll
                for (int mt = 0; mt < 4; ++mt) {
                    v16bf afrag = load_a(hp2, mt * 16 + mrow, kk + koffA);
                    acc[mt] = __builtin_amdgcn_wmma_f32_16x16x32_bf16(
                        false, afrag, false, bfrag, (short)0, acc[mt], false, false);
                }
            }
            const float bias = b2s[wv][mrow];
            #pragma unroll
            for (int mt = 0; mt < 4; ++mt)
                #pragma unroll
                for (int r = 0; r < 8; ++r) {
                    int bb = mt * 16 + r + bofs;
                    gLDS[wv][bb][mrow] = acc[mt][r] + bias;
                }
        }
        __syncthreads();
        for (int e = tid; e < BAT * 16; e += TPB) {
            int bb = e >> 4, j = e & 15;
            float i_ = gLDS[0][bb][j], f_ = gLDS[1][bb][j];
            float g_ = gLDS[2][bb][j], o_ = gLDS[3][bb][j];
            float c = f_ * cst[1][bb][j] + i_ * g_;
            cst[1][bb][j] = c;
            float h = o_ * tanhf(c);
            hn2[bb * HID + jb + j] = (bf16_t)h;
            htile[bb][j] = h;
        }
        __syncthreads();
        if (tid < BAT) {                                  // head partial for this slice
            float s = 0.f;
            #pragma unroll
            for (int j = 0; j < 16; ++j) s += htile[tid][j] * lws[j];
            part[blockIdx.x * BAT + tid] = s;
        }
        grid_barrier(bcnt, bgen);                         // h2_t + partials visible
    }

    // final output column
    if (blockIdx.x == 0 && tid < BAT) {
        float red = lbv;
        #pragma unroll 4
        for (int w = 0; w < NWG; ++w) red += part[w * BAT + tid];
        out[tid * T + (T - 1)] = red;
    }
}

// ---------------- host entry ----------------
extern "C" void kernel_launch(void* const* d_in, const int* in_sizes, int n_in,
                              void* d_out, int out_size, void* d_ws, size_t ws_size,
                              hipStream_t stream) {
    const float* x   = (const float*)d_in[0];
    const float* W1  = (const float*)d_in[1];
    const float* bW1 = (const float*)d_in[2];
    const float* U1  = (const float*)d_in[3];
    const float* bU1 = (const float*)d_in[4];
    const float* W2  = (const float*)d_in[5];
    const float* bW2 = (const float*)d_in[6];
    const float* U2  = (const float*)d_in[7];
    const float* bU2 = (const float*)d_in[8];
    const float* lw  = (const float*)d_in[9];
    const float* lb  = (const float*)d_in[10];
    const int*   fut = (const int*)d_in[11];

    const int SEQ = in_sizes[0] / BAT;
    unsigned char* ws = (unsigned char*)d_ws;

    const int prepN = (int)(3 * WMAT);
    prep_kernel<<<(prepN + 255) / 256, 256, 0, stream>>>(U1, W2, U2, ws);
    lstm_kernel<<<NWG, TPB, 0, stream>>>(x, W1, bW1, bU1, bW2, bU2, lw, lb, fut,
                                         (float*)d_out, ws, SEQ);
}
